// Attention_86423331930517
// MI455X (gfx1250) — matrix operand order repaired and checked by reading, not compiled, not run on hardware
//
#include <hip/hip_runtime.h>
#include <hip/hip_bf16.h>

// ---------------------------------------------------------------------------
// CDNA5 (gfx1250) WMMA types
// ---------------------------------------------------------------------------
typedef __attribute__((ext_vector_type(16))) __bf16 bf16x16;
typedef __attribute__((ext_vector_type(8)))  float  f32x8;

__device__ __forceinline__ f32x8 zero8() {
    f32x8 z = {0.f, 0.f, 0.f, 0.f, 0.f, 0.f, 0.f, 0.f};
    return z;
}

// float -> bf16 (round to nearest even), stored as raw u16
__device__ __forceinline__ unsigned short f2bf(float x) {
    union { float f; unsigned int u; } cv; cv.f = x;
    unsigned int r = cv.u + 0x7FFFu + ((cv.u >> 16) & 1u);
    return (unsigned short)(r >> 16);
}
__device__ __forceinline__ unsigned short to_bf16u(float x)          { return f2bf(x); }
__device__ __forceinline__ unsigned short to_bf16u(unsigned short x) { return x; }

// LDS byte offset of a __shared__ object (generic ptr low 32 bits = LDS offset)
__device__ __forceinline__ unsigned lds_off(const void* p) {
    return (unsigned)(unsigned long long)p;
}

// gfx1250 async global->LDS copy, 16B per lane, tracked on ASYNCcnt.
// LDS dest offset must be 16B aligned.
__device__ __forceinline__ void async_copy_b128(unsigned lds_byte_off, const void* gptr) {
    unsigned long long ga = (unsigned long long)gptr;
    asm volatile("global_load_async_to_lds_b128 %0, %1, off"
                 :: "v"(lds_byte_off), "v"(ga) : "memory");
}
__device__ __forceinline__ void wait_async() {
    asm volatile("s_wait_asynccnt 0" ::: "memory");
}

// single-instruction lane-XOR exchange: ds_swizzle group-of-32 mode
// imm: [4:0]=and_mask(0x1F) [9:5]=or_mask(0) [14:10]=xor_mask(K)
template <int K>
__device__ __forceinline__ float swz_xor(float v) {
    constexpr int imm = (K << 10) | 0x1F;
    union { float f; int i; } a, b;
    a.f = v;
    b.i = __builtin_amdgcn_ds_swizzle(a.i, imm);
    return b.f;
}

// ---------------------------------------------------------------------------
// Fragment loaders (ISA 7.12.2, 16-bit matrices, wave32).
//
// A (16xK slice, row m striped across VGPRs within a lane):
//   lane l: m = l&15, half = l>>4
//   vgpr j holds K = ((j<4)?0:16) + half*8 + (j&3)*2  (+0,+1)
//
// B (Kx16 slice) loaded from an LDS tile stored N-major ([n][k]):
//   lane l: n = l&15, half = l>>4
//   vgpr j holds K = ((j<4)?0:16) + half*8 + (j&3)*2  (+0,+1)
// ldk must be even so u32 loads of bf16 pairs stay 4B-aligned; with the
// strides used here each fragment folds into two ds_load_b128.
// ---------------------------------------------------------------------------
__device__ __forceinline__ bf16x16 load_frag_a(const unsigned short* base, int ldk, int lane) {
    const int m = lane & 15, half = lane >> 4;
    union { bf16x16 v; unsigned int u[8]; } f;
#pragma unroll
    for (int j = 0; j < 8; ++j) {
        const int k = ((j < 4) ? 0 : 16) + half * 8 + (j & 3) * 2;
        f.u[j] = *(const unsigned int*)(base + m * ldk + k);
    }
    return f.v;
}

__device__ __forceinline__ bf16x16 load_frag_b(const unsigned short* base, int ldk, int lane) {
    const int n = lane & 15, half = lane >> 4;
    union { bf16x16 v; unsigned int u[8]; } f;
#pragma unroll
    for (int j = 0; j < 8; ++j) {
        const int k = ((j < 4) ? 0 : 16) + half * 8 + (j & 3) * 2;
        f.u[j] = *(const unsigned int*)(base + n * ldk + k);
    }
    return f.v;
}

__device__ __forceinline__ f32x8 wmma_bf16(bf16x16 a, bf16x16 b, f32x8 c) {
    return __builtin_amdgcn_wmma_f32_16x16x32_bf16(false, a, false, b, (short)0, c, false, false);
}

// ---------------------------------------------------------------------------
// 2x2 average pool: x[16,64,64,512] f32 -> xp[16,32,32,512] f32
// ---------------------------------------------------------------------------
__global__ __launch_bounds__(256) void pool_kernel(const float* __restrict__ x,
                                                   float* __restrict__ xp) {
    const size_t i = (size_t)blockIdx.x * 256 + threadIdx.x;   // < 16*32*32*512
    const int c = (int)(i & 511);
    const int w = (int)((i >> 9)  & 31);
    const int h = (int)((i >> 14) & 31);
    const int b = (int)(i >> 19);
    const float* p = x + ((((size_t)b * 64 + 2 * h) * 64 + 2 * w) * 512 + c);
    xp[i] = 0.25f * (p[0] + p[512] + p[64 * 512] + p[64 * 512 + 512]);
}

// ---------------------------------------------------------------------------
// Generic tiled GEMM  C[M,N] = A[M,K] * Bw[K,N]   (bf16 WMMA, f32 accum)
//   AT = float (VALU-converted staging) or unsigned short (bf16: async B128
//        global->LDS staging, no VGPR round-trip)
//   OUT_F32_RES: write f32 with +Res residual, else write bf16
// BM=128 BN=64 BK=32, 256 threads = 8 waves, each wave owns a 16x64 strip.
// LDK=40: rows 80B (16B-aligned for async B128) and 20-dword bank stride
// (20*n mod 64 distinct for n=0..15 -> conflict-free fragment reads).
// ---------------------------------------------------------------------------
template <typename AT, bool OUT_F32_RES>
__global__ __launch_bounds__(256) void gemm_bf16_wmma(const AT* __restrict__ A,
                                                      const float* __restrict__ Bw,
                                                      void* __restrict__ Cout,
                                                      const float* __restrict__ Res,
                                                      int M, int N, int K) {
    constexpr int BM = 128, BN = 64, BK = 32, LDK = BK + 8;   // LDK = 40
    __shared__ __attribute__((aligned(16))) unsigned short As[BM * LDK];
    __shared__ __attribute__((aligned(16))) unsigned short Bs[BN * LDK];   // N-major: [n][k]

    const int tid  = threadIdx.x;
    const int lane = tid & 31;
    const int wave = tid >> 5;
    const int mBase = blockIdx.y * BM;
    const int nBase = blockIdx.x * BN;
    const unsigned AsOff = lds_off(&As[0]);

    f32x8 acc[4];
#pragma unroll
    for (int nt = 0; nt < 4; ++nt) acc[nt] = zero8();

    for (int k0 = 0; k0 < K; k0 += BK) {
        __syncthreads();
        // stage A tile
        if constexpr (sizeof(AT) == 2) {
            // bf16 source: async global->LDS, 512 x 16B chunks (4 per row)
#pragma unroll
            for (int i = 0; i < 2; ++i) {
                const int idx = tid + i * 256;
                const int r = idx >> 2, ch = idx & 3;
                async_copy_b128(AsOff + r * (LDK * 2) + ch * 16,
                                (const void*)(A + (size_t)(mBase + r) * K + k0 + ch * 8));
            }
        } else {
            // f32 source: load + convert (coalesced over K)
#pragma unroll
            for (int i = 0; i < 16; ++i) {
                const int idx = tid + i * 256;
                const int r = idx >> 5, c = idx & 31;
                As[r * LDK + c] = to_bf16u(A[(size_t)(mBase + r) * K + (k0 + c)]);
            }
        }
        // stage B tile transposed (coalesced over N)
#pragma unroll
        for (int i = 0; i < 8; ++i) {
            const int idx = tid + i * 256;
            const int kk = idx >> 6, n = idx & 63;
            Bs[n * LDK + kk] = f2bf(Bw[(size_t)(k0 + kk) * N + (nBase + n)]);
        }
        if constexpr (sizeof(AT) == 2) wait_async();
        __syncthreads();

        const bf16x16 af = load_frag_a(As + wave * 16 * LDK, LDK, lane);
#pragma unroll
        for (int nt = 0; nt < 4; ++nt) {
            const bf16x16 bf = load_frag_b(Bs + nt * 16 * LDK, LDK, lane);
            acc[nt] = wmma_bf16(af, bf, acc[nt]);
        }
    }

    // epilogue: C layout -> row = wave*16 + half*8 + r, col = nt*16 + (lane&15)
    const int half = lane >> 4, nl = lane & 15;
#pragma unroll
    for (int nt = 0; nt < 4; ++nt) {
#pragma unroll
        for (int r = 0; r < 8; ++r) {
            const int row = mBase + wave * 16 + half * 8 + r;
            const int col = nBase + nt * 16 + nl;
            const size_t o = (size_t)row * N + col;
            const float v = acc[nt][r];
            if constexpr (OUT_F32_RES) ((float*)Cout)[o] = v + Res[o];
            else                       ((unsigned short*)Cout)[o] = f2bf(v);
        }
    }
}

// ---------------------------------------------------------------------------
// Flash attention: per block = 64 queries of one batch (4 waves x 16 rows).
//   f[B,4096,64] bf16, g[B,1024,64] bf16, h[B,1024,256] bf16 -> y[B,4096,256] bf16
// Streams 16 key-blocks of 64; logits never touch HBM.
// LF=72: rows 144B (16B-aligned, async B128 staging for F/G) and 36-dword
// bank stride (36*n mod 64 distinct for n=0..15 -> conflict-free frags).
// ---------------------------------------------------------------------------
__global__ __launch_bounds__(128) void attn_flash(const unsigned short* __restrict__ fq,
                                                  const unsigned short* __restrict__ gk,
                                                  const unsigned short* __restrict__ hv,
                                                  unsigned short* __restrict__ y) {
    constexpr int D = 64, NK = 1024, NV = 256, QB = 64, KB = 64, LF = D + 8;  // LF = 72
    __shared__ __attribute__((aligned(16))) unsigned short Fs[QB * LF];       // [q][d]
    __shared__ __attribute__((aligned(16))) unsigned short Gs[KB * LF];       // [k][d]
    __shared__ __attribute__((aligned(16))) unsigned short Hs[NV * LF];       // [n][k]
    __shared__ __attribute__((aligned(16))) unsigned short Ps[4 * 16 * LF];   // per-wave [q][k]

    const int b  = blockIdx.x >> 6;
    const int qb = (blockIdx.x & 63) * QB;
    const int tid = threadIdx.x, lane = tid & 31, wave = tid >> 5;
    const int half = lane >> 4, nl = lane & 15;
    const unsigned FsOff = lds_off(&Fs[0]);
    const unsigned GsOff = lds_off(&Gs[0]);

    // async-stage F tile: 64 rows x 128B = 512 x 16B chunks
    const unsigned short* fbase = fq + ((size_t)b * 4096 + qb) * D;
#pragma unroll
    for (int i = 0; i < 4; ++i) {
        const int idx = tid + i * 128;
        const int q = idx >> 3, ch = idx & 7;
        async_copy_b128(FsOff + q * (LF * 2) + ch * 16, fbase + q * D + ch * 8);
    }

    f32x8 o[16];
#pragma unroll
    for (int nt = 0; nt < 16; ++nt) o[nt] = zero8();
    float mrow[8], lrow[8];
#pragma unroll
    for (int r = 0; r < 8; ++r) { mrow[r] = -3.0e38f; lrow[r] = 0.f; }

    for (int kb = 0; kb < NK; kb += KB) {
        __syncthreads();
        // async-stage G tile (pure bf16 copy)
        const unsigned short* gbase = gk + ((size_t)b * 1024 + kb) * D;
#pragma unroll
        for (int i = 0; i < 4; ++i) {
            const int idx = tid + i * 128;
            const int k = idx >> 3, ch = idx & 7;
            async_copy_b128(GsOff + k * (LF * 2) + ch * 16, gbase + k * D + ch * 8);
        }
        // stage H tile transposed into N-major (needs element shuffle)
        const unsigned short* hbase = hv + ((size_t)b * 1024 + kb) * NV;
        for (int idx = tid; idx < KB * NV; idx += 128) {
            const int k = idx >> 8, n = idx & 255;
            Hs[n * LF + k] = hbase[k * NV + n];
        }
        wait_async();
        __syncthreads();

        // S = F_w (16x64) * G^T  -> 4 accums of 16x16
        f32x8 s[4];
#pragma unroll
        for (int nt = 0; nt < 4; ++nt) s[nt] = zero8();
#pragma unroll
        for (int ks = 0; ks < D; ks += 32) {
            const bf16x16 af = load_frag_a(Fs + wave * 16 * LF + ks, LF, lane);
#pragma unroll
            for (int nt = 0; nt < 4; ++nt) {
                const bf16x16 bf = load_frag_b(Gs + nt * 16 * LF + ks, LF, lane);
                s[nt] = wmma_bf16(af, bf, s[nt]);
            }
        }

        // streaming softmax (row = half*8 + r lives across the 16 lanes of a half)
        // cross-lane reductions via single-instruction ds_swizzle XOR exchanges
        float scale[8];
#pragma unroll
        for (int r = 0; r < 8; ++r) {
            float m = fmaxf(fmaxf(s[0][r], s[1][r]), fmaxf(s[2][r], s[3][r]));
            m = fmaxf(m, swz_xor<8>(m));
            m = fmaxf(m, swz_xor<4>(m));
            m = fmaxf(m, swz_xor<2>(m));
            m = fmaxf(m, swz_xor<1>(m));
            const float mn = fmaxf(mrow[r], m);
            scale[r] = __expf(mrow[r] - mn);
            mrow[r] = mn;
            float psum = 0.f;
#pragma unroll
            for (int nt = 0; nt < 4; ++nt) {
                const float p = __expf(s[nt][r] - mn);
                s[nt][r] = p;
                psum += p;
            }
            psum += swz_xor<8>(psum);
            psum += swz_xor<4>(psum);
            psum += swz_xor<2>(psum);
            psum += swz_xor<1>(psum);
            lrow[r] = lrow[r] * scale[r] + psum;
        }
        // rescale running output
#pragma unroll
        for (int nt = 0; nt < 16; ++nt)
#pragma unroll
            for (int r = 0; r < 8; ++r) o[nt][r] *= scale[r];

        // stash P (C layout -> row-major) in this wave's LDS region
        unsigned short* pw = Ps + wave * 16 * LF;
#pragma unroll
        for (int nt = 0; nt < 4; ++nt)
#pragma unroll
            for (int r = 0; r < 8; ++r)
                pw[(half * 8 + r) * LF + nt * 16 + nl] = f2bf(s[nt][r]);
        asm volatile("s_wait_dscnt 0" ::: "memory");   // same-wave LDS RAW

        // O += P (16x64) * H (64x256)
#pragma unroll
        for (int ks = 0; ks < KB; ks += 32) {
            const bf16x16 af = load_frag_a(pw + ks, LF, lane);
#pragma unroll
            for (int nt = 0; nt < 16; ++nt) {
                const bf16x16 bf = load_frag_b(Hs + nt * 16 * LF + ks, LF, lane);
                o[nt] = wmma_bf16(af, bf, o[nt]);
            }
        }
    }

    // epilogue: y = O / l
    unsigned short* ybase = y + ((size_t)b * 4096 + qb) * NV;
    float inv[8];
#pragma unroll
    for (int r = 0; r < 8; ++r) inv[r] = 1.f / lrow[r];
#pragma unroll
    for (int nt = 0; nt < 16; ++nt)
#pragma unroll
        for (int r = 0; r < 8; ++r) {
            const int row = wave * 16 + half * 8 + r;
            const int col = nt * 16 + nl;
            ybase[(size_t)row * NV + col] = f2bf(o[nt][r] * inv[r]);
        }
}

// ---------------------------------------------------------------------------
// Launch
// ---------------------------------------------------------------------------
extern "C" void kernel_launch(void* const* d_in, const int* in_sizes, int n_in,
                              void* d_out, int out_size, void* d_ws, size_t ws_size,
                              hipStream_t stream) {
    (void)in_sizes; (void)n_in; (void)out_size; (void)ws_size;
    const float* x  = (const float*)d_in[0];   // [16,64,64,512]
    const float* Wf = (const float*)d_in[1];   // [512,64]
    const float* Wg = (const float*)d_in[2];   // [512,64]
    const float* Wh = (const float*)d_in[3];   // [512,256]
    const float* Wo = (const float*)d_in[4];   // [256,512]
    float* out = (float*)d_out;                // [16,64,64,512]

    constexpr int B = 16, HW = 4096, PHW = 1024, C = 512, DQ = 64, DV = 256;
    constexpr size_t M_Q = (size_t)B * HW;     // 65536
    constexpr size_t M_P = (size_t)B * PHW;    // 16384

    // workspace carve-up (256B aligned)
    char* ws = (char*)d_ws;
    size_t off = 0;
    auto carve = [&](size_t bytes) { char* p = ws + off; off = (off + bytes + 255) & ~(size_t)255; return p; };
    float*          xp  = (float*)         carve(M_P * C  * sizeof(float));          // 33.6 MB
    unsigned short* fws = (unsigned short*)carve(M_Q * DQ * sizeof(unsigned short)); //  8.4 MB
    unsigned short* gws = (unsigned short*)carve(M_P * DQ * sizeof(unsigned short)); //  2.1 MB
    unsigned short* hws = (unsigned short*)carve(M_P * DV * sizeof(unsigned short)); //  8.4 MB
    unsigned short* yws = (unsigned short*)carve(M_Q * DV * sizeof(unsigned short)); // 33.6 MB

    // 1) 2x2 average pool
    pool_kernel<<<(int)(M_P * C / 256), 256, 0, stream>>>(x, xp);

    // 2) projections: f = x*Wf, g = xp*Wg, h = xp*Wh  (bf16 outputs)
    gemm_bf16_wmma<float, false><<<dim3(DQ / 64, M_Q / 128), 256, 0, stream>>>(
        x,  Wf, fws, nullptr, (int)M_Q, DQ, C);
    gemm_bf16_wmma<float, false><<<dim3(DQ / 64, M_P / 128), 256, 0, stream>>>(
        xp, Wg, gws, nullptr, (int)M_P, DQ, C);
    gemm_bf16_wmma<float, false><<<dim3(DV / 64, M_P / 128), 256, 0, stream>>>(
        xp, Wh, hws, nullptr, (int)M_P, DV, C);

    // 3) flash attention: y = softmax(f g^T) h
    attn_flash<<<B * (HW / 64), 128, 0, stream>>>(fws, gws, hws, yws);

    // 4) out = y*Wo + x  (f32 output with residual)
    gemm_bf16_wmma<unsigned short, true><<<dim3(C / 64, M_Q / 128), 256, 0, stream>>>(
        yws, Wo, out, x, (int)M_Q, C, DV);
}
